// RED_Regularization_51599737094654
// MI455X (gfx1250) — matrix operand order repaired and checked by reading, not compiled, not run on hardware
//
#include <hip/hip_runtime.h>
#include <hip/hip_bf16.h>

typedef __attribute__((ext_vector_type(16))) __bf16 v16bf;
typedef __attribute__((ext_vector_type(8)))  __bf16 v8bf;
typedef __attribute__((ext_vector_type(8)))  float  v8f;
typedef __attribute__((ext_vector_type(4)))  unsigned int u32x4;
typedef __attribute__((ext_vector_type(8)))  int i32x8;
typedef __attribute__((ext_vector_type(4)))  int i32x4;

// Tensor Data Mover builtin: 5-arg form on ROCm 7.2 (clang-22), 6-arg form on
// the clang-23 toolchain. Manual copy as final fallback.
#if defined(__HIP_DEVICE_COMPILE__) &&                    \
    __has_builtin(__builtin_amdgcn_tensor_load_to_lds) && \
    __has_builtin(__builtin_amdgcn_s_wait_tensorcnt)
#define RED_HAVE_TDM 1
#if __clang_major__ >= 23
#define RED_TDM6 1
#else
#define RED_TDM6 0
#endif
#else
#define RED_HAVE_TDM 0
#define RED_TDM6 0
#endif

namespace {
constexpr int H0 = 256, W0 = 320, DD = 48, CIN = 8;
constexpr int OP_LIN = 0, OP_RELU = 1, OP_SIG = 2, OP_TANH = 3;
// 2-bit packed ky/kx lookup for taps 0..8 (tap = 3*ky + kx)
constexpr unsigned KYC = 0x2A540u;   // ky: 0,0,0,1,1,1,2,2,2
constexpr unsigned KXC = 0x24924u;   // kx: 0,1,2,0,1,2,0,1,2
}

// ---------------------------------------------------------------------------
// Slice d of the NCHW fp32 cost volume -> negated bf16 channels-last, written
// into the interior of a zero-haloed [(H+2)x(W+2)][16] buffer.
// ---------------------------------------------------------------------------
__global__ void k_slice_neg(const float* __restrict__ vv, int d,
                            __bf16* __restrict__ dst) {
  int p = blockIdx.x * blockDim.x + threadIdx.x;
  if (p >= H0 * W0) return;
  int h = p / W0, w = p - h * W0;
  v16bf o;
#pragma unroll
  for (int c = 0; c < 8; ++c) o[c] = (__bf16)(-vv[(c * DD + d) * (H0 * W0) + p]);
#pragma unroll
  for (int c = 8; c < 16; ++c) o[c] = (__bf16)0.0f;
  *(v16bf*)(dst + (size_t)((h + 1) * (W0 + 2) + w + 1) * 16) = o;
}

// ---------------------------------------------------------------------------
// Pre-swizzle OIHW fp32 weights into WMMA A-fragment layout with taps packed
// into K: global K index g = tap*CS + c, chunks of 32.
// Layout: [mg][kc][mb][lane(32)][elem(16)], m = (mg*MB+mb)*16 + (lane&15)
// ---------------------------------------------------------------------------
__global__ void k_prep_w(const float* __restrict__ w, int Cout, int Cin,
                         int CS, int lgCS, __bf16* __restrict__ dst, int MB,
                         int KCtot, int total) {
  int idx = blockIdx.x * blockDim.x + threadIdx.x;
  if (idx >= total) return;
  int e = idx & 15;
  int lane = (idx >> 4) & 31;
  int t = idx >> 9;
  int mb = t % MB; t /= MB;
  int kc = t % KCtot; t /= KCtot;
  int mg = t;
  int m = (mg * MB + mb) * 16 + (lane & 15);
  int klocal = e + ((lane >= 16) ? 16 : 0);
  int g = kc * 32 + klocal;
  int tap = g >> lgCS;
  int c = g & (CS - 1);
  float v = 0.0f;
  if (tap < 9 && m < Cout && c < Cin) {
    int ky = tap / 3, kx = tap - ky * 3;
    v = w[((m * Cin + c) * 3 + ky) * 3 + kx];
  }
  dst[idx] = (__bf16)v;
}

// ---------------------------------------------------------------------------
// Implicit-GEMM 3x3 conv (pad=1) via v_wmma_f32_16x16x32_bf16.
// One wave: 16 pixels x (MB*16) output channels; K = 9 taps x CS channels
// packed contiguously. Zero-halo inputs -> unconditional B loads (invalid
// taps select offset 0, a guaranteed-zero border pixel). A and B fragments
// double-buffered one chunk ahead. A slab staged into LDS per workgroup via
// one 1-D Tensor Data Mover transfer when it fits.
// TR=1: input-dilated (ConvTranspose k3 s2 p1 op1) coordinate mapping.
// ---------------------------------------------------------------------------
template <int MB, int TR>
__global__ __launch_bounds__(256) void k_conv3x3_wmma(
    const __bf16* __restrict__ in, int Hin, int Win, int lgCS,
    __bf16* __restrict__ outp, int Hout, int Wout, int Cout, int CSout,
    const __bf16* __restrict__ wprep, const float* __restrict__ bias,
    int KCtot, int stride, int op, int use_lds, float* __restrict__ out_f32) {
  extern __shared__ char dynsmem[];
  __bf16* aslab = (__bf16*)dynsmem;
  const int mg = blockIdx.y;
  const int slabElems = KCtot * MB * 512;
  const __bf16* gslab = wprep + (size_t)mg * slabElems;
  if (use_lds) {
#if RED_HAVE_TDM
    if (threadIdx.x < 32) {  // one wave issues the TDM descriptor
      const unsigned n8 = (unsigned)(slabElems >> 2);  // bytes/8 (<= 6144)
      unsigned lds_lo = (unsigned)(size_t)aslab;
      unsigned long long ga = (unsigned long long)(size_t)gslab;
      u32x4 g0;
      g0[0] = 1u;                                  // count=1 user descriptor
      g0[1] = lds_lo;                              // lds_addr
      g0[2] = (unsigned)ga;                        // global_addr[31:0]
      g0[3] = (unsigned)((ga >> 32) & 0x01FFFFFFu) | (2u << 30);  // hi|type=2
      i32x8 g1;
      g1[0] = (int)(3u << 16);                     // data_size=8B, mask=0
      g1[1] = (int)((n8 & 0xFFFFu) << 16);         // tensor_dim0 lo16 @bit48
      g1[2] = (int)(((n8 >> 16) & 0xFFFFu) | (1u << 16));  // dim0 hi|dim1=1
      g1[3] = (int)((n8 & 0xFFFFu) << 16);         // tile_dim0 = whole tensor
      g1[4] = 1;                                   // tile_dim1 = 1
      g1[5] = (int)n8;                             // tensor_dim0_stride lo32
      g1[6] = 0; g1[7] = 0;
      i32x4 gz;
      gz[0] = 0; gz[1] = 0; gz[2] = 0; gz[3] = 0;
#if RED_TDM6
      i32x8 gz8;
      gz8[0] = 0; gz8[1] = 0; gz8[2] = 0; gz8[3] = 0;
      gz8[4] = 0; gz8[5] = 0; gz8[6] = 0; gz8[7] = 0;
      __builtin_amdgcn_tensor_load_to_lds(g0, g1, gz, gz, gz8, 0);
#else
      __builtin_amdgcn_tensor_load_to_lds(g0, g1, gz, gz, 0);
#endif
      __builtin_amdgcn_s_wait_tensorcnt(0);
    }
#else
    const v8bf* s = (const v8bf*)gslab;
    v8bf* dsts = (v8bf*)aslab;
    for (int i = threadIdx.x; i < (slabElems >> 3); i += 256) dsts[i] = s[i];
#endif
  }
  __syncthreads();

  const int lane = threadIdx.x & 31;
  const int HWo = Hout * Wout;
  const int NT = (HWo + 15) >> 4;
  const int nt = blockIdx.x * 8 + (threadIdx.x >> 5);
  if (nt >= NT) return;
  const int n = nt * 16 + (lane & 15);
  const bool npix = n < HWo;
  int oh = 0, ow = 0;
  if (npix) { oh = n / Wout; ow = n - oh * Wout; }
  const int half = lane >> 4;
  const int koff = half * 16;
  const int cmask = (1 << lgCS) - 1;
  const int Wp = Win + 2;                 // padded input row pitch (pixels)
  const int WpCS = Wp << lgCS;
  const int base0 = TR ? 0 : (((oh * stride) * Wp + ow * stride) << lgCS);

  auto loadB = [&](int kc) -> v16bf {
    int g0 = kc * 32 + koff;
    int tap = g0 >> lgCS;                 // which 3x3 tap this half-chunk is
    int c0 = g0 & cmask;                  // channel offset within the tap
    int t2 = tap << 1;
    int ky = (KYC >> t2) & 3;
    int kx = (KXC >> t2) & 3;
    bool valid = npix && (tap < 9);
    int off;
    if constexpr (TR) {
      int da = oh + ky - 1, db = ow + kx - 1;   // dilated coords, must be even
      valid = valid && (((da | db) & 1) == 0);
      off = ((((da >> 1) + 1) * Wp + (db >> 1) + 1) << lgCS);
    } else {
      off = base0 + ky * WpCS + (kx << lgCS);   // halo absorbs the -1 pad
    }
    off = valid ? off : 0;                // border pixel 0 is all zeros
    unsigned ob = (unsigned)(off + c0) * 2u;  // 32-bit byte offset -> SADDR form
    return *(const v16bf*)((const char*)in + (size_t)ob);
  };
  auto loadA = [&](const __bf16* __restrict__ ab, int kc, int mb) -> v16bf {
    return *(const v16bf*)(ab + ((kc * MB + mb) * 32 + lane) * 16);
  };

  v8f acc[MB];
#pragma unroll
  for (int mb = 0; mb < MB; ++mb)
#pragma unroll
    for (int i = 0; i < 8; ++i) acc[mb][i] = 0.0f;

  auto body = [&](const __bf16* __restrict__ ab) {
    v16bf bcur = loadB(0);
    v16bf acur[MB];
#pragma unroll
    for (int mb = 0; mb < MB; ++mb) acur[mb] = loadA(ab, 0, mb);
    for (int kc = 0; kc < KCtot - 1; ++kc) {
      v16bf bnxt = loadB(kc + 1);         // prefetch next chunk's B
      v16bf anxt[MB];
#pragma unroll
      for (int mb = 0; mb < MB; ++mb) anxt[mb] = loadA(ab, kc + 1, mb);
#pragma unroll
      for (int mb = 0; mb < MB; ++mb)
        acc[mb] = __builtin_amdgcn_wmma_f32_16x16x32_bf16(
            false, acur[mb], false, bcur, (short)0, acc[mb], false, false);
      bcur = bnxt;
#pragma unroll
      for (int mb = 0; mb < MB; ++mb) acur[mb] = anxt[mb];
    }
#pragma unroll
    for (int mb = 0; mb < MB; ++mb)
      acc[mb] = __builtin_amdgcn_wmma_f32_16x16x32_bf16(
          false, acur[mb], false, bcur, (short)0, acc[mb], false, false);
  };
  if (use_lds) body(aslab); else body(gslab);

  if (out_f32) {  // single-channel fp32 output (final regularization map)
    if (npix && half == 0 && mg == 0) out_f32[n] = acc[0][0] + bias[0];
    return;
  }
  if (!npix) return;
  const int Wpo = Wout + 2;
  const int pout = (oh + 1) * Wpo + ow + 1;   // interior of haloed output
#pragma unroll
  for (int mb = 0; mb < MB; ++mb) {
    int m0 = (mg * MB + mb) * 16 + half * 8;  // C/D: VGPR r -> M=r+half*8
    v8bf res;
#pragma unroll
    for (int r = 0; r < 8; ++r) {
      int m = m0 + r;
      float v = 0.0f;
      if (m < Cout) {
        v = acc[mb][r] + bias[m];
        if (op == OP_RELU) v = v > 0.0f ? v : 0.0f;
        else if (op == OP_SIG) v = 1.0f / (1.0f + __expf(-v));
        else if (op == OP_TANH) v = tanhf(v);
      }
      res[r] = (__bf16)v;  // pad channels stay exactly 0
    }
    *(v8bf*)(outp + (size_t)pout * CSout + m0) = res;
  }
}

// ---------------------------------------------------------------------------
// Elementwise helpers, vectorized 8-wide, run over the FULL haloed extent
// (zero inputs at borders map to zero outputs, halo stays zero).
// ---------------------------------------------------------------------------
__global__ void k_concat(const __bf16* __restrict__ a, int C1, int CS1,
                         const __bf16* __restrict__ b, int C2, int CS2,
                         __bf16* __restrict__ dst, int CSd, int HW) {
  int idx = blockIdx.x * blockDim.x + threadIdx.x;
  int gpp = CSd >> 3;
  if (idx >= HW * gpp) return;
  int c8 = (idx % gpp) << 3;
  int p = idx / gpp;
  v8bf r;
  if (c8 < C1) {
    r = *(const v8bf*)(a + p * CS1 + c8);
  } else if (c8 < C1 + C2) {
    r = *(const v8bf*)(b + p * CS2 + (c8 - C1));
  } else {
#pragma unroll
    for (int i = 0; i < 8; ++i) r[i] = (__bf16)0.0f;
  }
  *(v8bf*)(dst + p * CSd + c8) = r;
}

__global__ void k_gate_mul(const __bf16* __restrict__ gates, int CSg,
                           const __bf16* __restrict__ h, int CSh, int Ch,
                           __bf16* __restrict__ dst, int HW) {
  int idx = blockIdx.x * blockDim.x + threadIdx.x;
  int gpp = CSh >> 3;
  if (idx >= HW * gpp) return;
  int c8 = (idx % gpp) << 3;
  int p = idx / gpp;
  v8bf r;
  if (c8 < Ch) {
    v8bf g = *(const v8bf*)(gates + p * CSg + c8);
    v8bf hh = *(const v8bf*)(h + p * CSh + c8);
#pragma unroll
    for (int i = 0; i < 8; ++i) r[i] = (__bf16)((float)g[i] * (float)hh[i]);
  } else {
#pragma unroll
    for (int i = 0; i < 8; ++i) r[i] = (__bf16)0.0f;
  }
  *(v8bf*)(dst + p * CSh + c8) = r;
}

__global__ void k_gru_combine(const __bf16* __restrict__ gates, int CSg, int Ch,
                              const __bf16* __restrict__ h,
                              const __bf16* __restrict__ cand,
                              __bf16* __restrict__ dst, int CSh, int HW) {
  int idx = blockIdx.x * blockDim.x + threadIdx.x;
  int gpp = CSh >> 3;
  if (idx >= HW * gpp) return;
  int c8 = (idx % gpp) << 3;
  int p = idx / gpp;
  v8bf r;
  if (c8 < Ch) {
    v8bf u = *(const v8bf*)(gates + p * CSg + Ch + c8);
    v8bf hh = *(const v8bf*)(h + p * CSh + c8);
    v8bf cc = *(const v8bf*)(cand + p * CSh + c8);
#pragma unroll
    for (int i = 0; i < 8; ++i) {
      float uf = (float)u[i];
      r[i] = (__bf16)(uf * (float)hh[i] + (1.0f - uf) * (float)cc[i]);
    }
  } else {
#pragma unroll
    for (int i = 0; i < 8; ++i) r[i] = (__bf16)0.0f;
  }
  *(v8bf*)(dst + p * CSh + c8) = r;
}

__global__ void k_add(const __bf16* __restrict__ a, const __bf16* __restrict__ b,
                      __bf16* __restrict__ dst, int ngroups) {
  int idx = blockIdx.x * blockDim.x + threadIdx.x;
  if (idx >= ngroups) return;
  v8bf x = ((const v8bf*)a)[idx];
  v8bf y = ((const v8bf*)b)[idx];
  v8bf r;
#pragma unroll
  for (int i = 0; i < 8; ++i) r[i] = (__bf16)((float)x[i] + (float)y[i]);
  ((v8bf*)dst)[idx] = r;
}

// ---------------------------------------------------------------------------
// Host orchestration
// ---------------------------------------------------------------------------
extern "C" void kernel_launch(void* const* d_in, const int* in_sizes, int n_in,
                              void* d_out, int out_size, void* d_ws, size_t ws_size,
                              hipStream_t stream) {
  (void)in_sizes; (void)n_in; (void)out_size; (void)ws_size;
  const float* vv = (const float*)d_in[0];
  float* out = (float*)d_out;

  struct CDesc { int wi, bi, Cout, Cin; };
  static const CDesc cd[15] = {
      {1, 2, 16, 8},    {3, 4, 32, 16},   {5, 6, 64, 32},    // conv1..3
      {7, 8, 32, 64},   {9, 10, 16, 32},  {11, 12, 8, 16},   // up3..1
      {13, 14, 1, 8},                                        // out
      {15, 16, 16, 16}, {17, 18, 8, 16},                     // g1 gate/cand
      {19, 20, 32, 32}, {21, 22, 16, 32},                    // g2
      {23, 24, 64, 64}, {25, 26, 32, 64},                    // g3
      {27, 28, 128, 128}, {29, 30, 64, 128}};                // g4

  struct CInfo { int MT, MB, MG, CS, lgCS, KCtot, use_lds; };
  CInfo ci[15];
  for (int i = 0; i < 15; ++i) {
    int MT = (cd[i].Cout + 15) / 16;
    int CS = (cd[i].Cin + 15) & ~15;
    int lg = (CS == 16) ? 4 : (CS == 32) ? 5 : (CS == 64) ? 6 : 7;
    int KC = (9 * CS + 31) / 32;
    int MB = MT >= 4 ? 4 : MT;           // m-tiles per wave (B-frag reuse)
    ci[i] = {MT, MB, MT / MB, CS, lg, KC,
             (KC * MB * 1024 <= 48 * 1024) ? 1 : 0};
  }

  size_t off = 0;
  char* base = (char*)d_ws;
  auto alloc = [&](size_t elems) -> __bf16* {
    off = (off + 255) & ~(size_t)255;
    __bf16* p = (__bf16*)(base + off);
    off += elems * sizeof(__bf16);
    return p;
  };

  __bf16* wp[15];
  for (int i = 0; i < 15; ++i)
    wp[i] = alloc((size_t)ci[i].MT * ci[i].KCtot * 512);

  // haloed pixel counts per level
  const int P0 = (H0 + 2) * (W0 + 2);        // 258*322
  const int P1 = (128 + 2) * (160 + 2);      // 130*162
  const int P2 = (64 + 2) * (80 + 2);        // 66*82
  const int P3 = (32 + 2) * (40 + 2);        // 34*42
  const int HW = H0 * W0;

  size_t actStart = (off + 255) & ~(size_t)255;
  __bf16* x_in = alloc((size_t)P0 * 16);
  __bf16* c1 = alloc((size_t)P1 * 16);
  __bf16* c2 = alloc((size_t)P2 * 32);
  __bf16* c3 = alloc((size_t)P3 * 64);
  __bf16* s1[2] = {alloc((size_t)P0 * 16), alloc((size_t)P0 * 16)};
  __bf16* s2[2] = {alloc((size_t)P1 * 16), alloc((size_t)P1 * 16)};
  __bf16* s3[2] = {alloc((size_t)P2 * 32), alloc((size_t)P2 * 32)};
  __bf16* s4[2] = {alloc((size_t)P3 * 64), alloc((size_t)P3 * 64)};
  __bf16* cat1 = alloc((size_t)P0 * 16);  __bf16* gates1 = alloc((size_t)P0 * 16);
  __bf16* rh1 = alloc((size_t)P0 * 16);   __bf16* cand1 = alloc((size_t)P0 * 16);
  __bf16* cat2 = alloc((size_t)P1 * 32);  __bf16* gates2 = alloc((size_t)P1 * 32);
  __bf16* rh2 = alloc((size_t)P1 * 16);   __bf16* cand2 = alloc((size_t)P1 * 16);
  __bf16* cat3 = alloc((size_t)P2 * 64);  __bf16* gates3 = alloc((size_t)P2 * 64);
  __bf16* rh3 = alloc((size_t)P2 * 32);   __bf16* cand3 = alloc((size_t)P2 * 32);
  __bf16* cat4 = alloc((size_t)P3 * 128); __bf16* gates4 = alloc((size_t)P3 * 128);
  __bf16* rh4 = alloc((size_t)P3 * 64);   __bf16* cand4 = alloc((size_t)P3 * 64);
  __bf16* u3 = alloc((size_t)P2 * 32);    __bf16* sum3 = alloc((size_t)P2 * 32);
  __bf16* u2 = alloc((size_t)P1 * 16);    __bf16* sum2 = alloc((size_t)P1 * 16);
  __bf16* u1 = alloc((size_t)P0 * 16);    __bf16* sum1 = alloc((size_t)P0 * 16);
  size_t actEnd = off;

  auto nb = [](long n) { return (int)((n + 255) / 256); };

  for (int i = 0; i < 15; ++i) {
    int total = ci[i].MT * ci[i].KCtot * 512;
    k_prep_w<<<nb(total), 256, 0, stream>>>(
        (const float*)d_in[cd[i].wi], cd[i].Cout, cd[i].Cin, ci[i].CS,
        ci[i].lgCS, wp[i], ci[i].MB, ci[i].KCtot, total);
  }
  // Zero all activation buffers: establishes zero halos and zero GRU states.
  (void)hipMemsetAsync(base + actStart, 0, actEnd - actStart, stream);

  auto conv = [&](const __bf16* in, int Hin, int Win, __bf16* o, int Hout,
                  int Wout, int Cout, int CSout, int k, int stride,
                  int transposed, int op, float* of32) {
    const CInfo& I = ci[k];
    int NT = (Hout * Wout + 15) / 16;
    dim3 grid((NT + 7) / 8, I.MG);
    size_t smem = I.use_lds ? (size_t)I.KCtot * I.MB * 1024 : 0;
    const float* bias = (const float*)d_in[cd[k].bi];
#define LAUNCH(MBV, TRV)                                                     \
  k_conv3x3_wmma<MBV, TRV><<<grid, 256, smem, stream>>>(                     \
      in, Hin, Win, I.lgCS, o, Hout, Wout, Cout, CSout, wp[k], bias,         \
      I.KCtot, stride, op, I.use_lds, of32)
    if (transposed) {
      switch (I.MB) {
        case 1: LAUNCH(1, 1); break;
        case 2: LAUNCH(2, 1); break;
        default: LAUNCH(4, 1); break;
      }
    } else {
      switch (I.MB) {
        case 1: LAUNCH(1, 0); break;
        case 2: LAUNCH(2, 0); break;
        default: LAUNCH(4, 0); break;
      }
    }
#undef LAUNCH
  };

  auto gru = [&](const __bf16* x, int Cx, int CSx, __bf16* hc, __bf16* hn,
                 int Ch, int CSh, int Hl, int Wl, int Pl, __bf16* cat,
                 int CScat, __bf16* gates, int CSg, __bf16* rh, __bf16* cand,
                 int gci, int cci) {
    k_concat<<<nb((long)Pl * CScat / 8), 256, 0, stream>>>(
        x, Cx, CSx, hc, Ch, CSh, cat, CScat, Pl);
    conv(cat, Hl, Wl, gates, Hl, Wl, 2 * Ch, CSg, gci, 1, 0, OP_SIG, nullptr);
    k_gate_mul<<<nb((long)Pl * CSh / 8), 256, 0, stream>>>(gates, CSg, hc, CSh,
                                                           Ch, rh, Pl);
    k_concat<<<nb((long)Pl * CScat / 8), 256, 0, stream>>>(
        x, Cx, CSx, rh, Ch, CSh, cat, CScat, Pl);
    conv(cat, Hl, Wl, cand, Hl, Wl, Ch, CSh, cci, 1, 0, OP_TANH, nullptr);
    k_gru_combine<<<nb((long)Pl * CSh / 8), 256, 0, stream>>>(
        gates, CSg, Ch, hc, cand, hn, CSh, Pl);
  };

  int cur = 0;
  for (int d = 0; d < DD; ++d) {
    int nxt = cur ^ 1;
    k_slice_neg<<<nb(HW), 256, 0, stream>>>(vv, d, x_in);
    conv(x_in, 256, 320, c1, 128, 160, 16, 16, 0, 2, 0, OP_RELU, nullptr);
    conv(c1, 128, 160, c2, 64, 80, 32, 32, 1, 2, 0, OP_RELU, nullptr);
    conv(c2, 64, 80, c3, 32, 40, 64, 64, 2, 2, 0, OP_RELU, nullptr);
    gru(c3, 64, 64, s4[cur], s4[nxt], 64, 64, 32, 40, P3, cat4, 128, gates4,
        128, rh4, cand4, 13, 14);
    conv(s4[nxt], 32, 40, u3, 64, 80, 32, 32, 3, 1, 1, OP_RELU, nullptr);
    gru(c2, 32, 32, s3[cur], s3[nxt], 32, 32, 64, 80, P2, cat3, 64, gates3, 64,
        rh3, cand3, 11, 12);
    k_add<<<nb((long)P2 * 32 / 8), 256, 0, stream>>>(u3, s3[nxt], sum3,
                                                     P2 * 32 / 8);
    conv(sum3, 64, 80, u2, 128, 160, 16, 16, 4, 1, 1, OP_RELU, nullptr);
    gru(c1, 16, 16, s2[cur], s2[nxt], 16, 16, 128, 160, P1, cat2, 32, gates2,
        32, rh2, cand2, 9, 10);
    k_add<<<nb((long)P1 * 16 / 8), 256, 0, stream>>>(u2, s2[nxt], sum2,
                                                     P1 * 16 / 8);
    conv(sum2, 128, 160, u1, 256, 320, 8, 16, 5, 1, 1, OP_RELU, nullptr);
    gru(x_in, 8, 16, s1[cur], s1[nxt], 8, 16, 256, 320, P0, cat1, 16, gates1,
        16, rh1, cand1, 7, 8);
    k_add<<<nb((long)P0 * 16 / 8), 256, 0, stream>>>(u1, s1[nxt], sum1,
                                                     P0 * 16 / 8);
    conv(sum1, 256, 320, gates1 /*unused bf16 dst*/, 256, 320, 1, 16, 6, 1, 0,
         OP_LIN, out + (size_t)d * HW);
    cur = nxt;
  }
}
